// TransformerEncoder_5171140624933
// MI455X (gfx1250) — matrix operand order repaired and checked by reading, not compiled, not run on hardware
//
#include <hip/hip_runtime.h>
#include <hip/hip_bf16.h>
#include <math.h>

// ---- problem constants ----
#define B_    2
#define L_    1024
#define OBS_  64
#define ACT_  16
#define OE_   512
#define AE_   256
#define RE_   128
#define TE_   128
#define C_    1024
#define H_    16
#define NL_   4
#define FF_   4096
#define DH_   64
#define MTOK  (B_*L_)          // 2048 tokens

typedef __attribute__((ext_vector_type(16))) __bf16 v16bf;
typedef __attribute__((ext_vector_type(8)))  float  v8f;

__device__ __forceinline__ unsigned short f2bf(float f) {
    union { float f; unsigned u; } x; x.f = f;
    unsigned r = x.u + 0x7FFFu + ((x.u >> 16) & 1u);
    return (unsigned short)(r >> 16);
}
__device__ __forceinline__ v8f vzero() {
    v8f z;
    #pragma unroll
    for (int i = 0; i < 8; i++) z[i] = 0.f;
    return z;
}

union Frag { v16bf v; unsigned u[8]; };

// LDS byte offset of a __shared__ object (generic -> AS(3) -> int)
#define LDS_OFF(p) ((unsigned)(unsigned long long)(__attribute__((address_space(3))) const void*)(const void*)(p))

// CDNA5 async global->LDS copy (16B per lane), tracked by ASYNCcnt
__device__ __forceinline__ void async_copy_b128(unsigned ldsoff, const void* gaddr) {
    asm volatile("global_load_async_to_lds_b128 %0, %1, off"
                 :: "v"(ldsoff), "v"(gaddr) : "memory");
}
__device__ __forceinline__ void wait_async0() {
    asm volatile("s_wait_asynccnt 0x0" ::: "memory");
}

// ------------------------------------------------------------------
// batchnorm stats over tran = obs[l]-obs[l-1] (tran[.,0]=0), per feature
// ------------------------------------------------------------------
__global__ void bn_stats_kernel(const float* __restrict__ obs, float* __restrict__ stats) {
    int f = blockIdx.x, tid = threadIdx.x;
    float s = 0.f, s2 = 0.f;
    for (int t = tid; t < MTOK; t += 256) {
        int b = t / L_, l = t % L_;
        float v = 0.f;
        if (l > 0) v = obs[(b*L_ + l)*OBS_ + f] - obs[(b*L_ + l - 1)*OBS_ + f];
        s += v; s2 += v*v;
    }
    __shared__ float sh[512];
    sh[tid] = s; sh[256 + tid] = s2; __syncthreads();
    for (int o = 128; o > 0; o >>= 1) {
        if (tid < o) { sh[tid] += sh[tid + o]; sh[256 + tid] += sh[256 + tid + o]; }
        __syncthreads();
    }
    if (tid == 0) {
        float m = sh[0] / MTOK;
        float v = sh[256] / MTOK - m*m;
        stats[f] = m; stats[64 + f] = v;
    }
}

// ------------------------------------------------------------------
// token encoder: concat of 4 small projections -> x fp32 [MTOK][C]
// ------------------------------------------------------------------
__global__ void encode_kernel(const float* __restrict__ obs, const float* __restrict__ act,
                              const float* __restrict__ rew,
                              const float* __restrict__ obsW, const float* __restrict__ obsb,
                              const float* __restrict__ actW, const float* __restrict__ actb,
                              const float* __restrict__ rewW, const float* __restrict__ rewb,
                              const float* __restrict__ trW,  const float* __restrict__ trb,
                              const float* __restrict__ gamma, const float* __restrict__ beta,
                              const float* __restrict__ stats, float* __restrict__ x) {
    int tok = blockIdx.x;
    int col = blockIdx.y*256 + threadIdx.x;
    int b = tok / L_, l = tok % L_;
    const float* prev = obs + ((long)b*L_ + (l > 0 ? l - 1 : 0))*OBS_;   // obs2 row
    float acc;
    if (col < OE_) {
        acc = obsb[col];
        for (int k = 0; k < OBS_; k++) acc += prev[k]*obsW[k*OE_ + col];
    } else if (col < OE_ + AE_) {
        int c = col - OE_;
        acc = actb[c];
        const float* a = act + (long)tok*ACT_;
        for (int k = 0; k < ACT_; k++) acc += a[k]*actW[k*AE_ + c];
    } else if (col < OE_ + AE_ + RE_) {
        int c = col - OE_ - AE_;
        acc = rewb[c] + rew[tok]*rewW[c];
    } else {
        int c = col - OE_ - AE_ - RE_;
        acc = trb[c];
        const float* cur = obs + (long)tok*OBS_;
        for (int k = 0; k < OBS_; k++) {
            float tr = (l > 0) ? (cur[k] - prev[k]) : 0.f;
            float tn = (tr - stats[k]) * rsqrtf(stats[64 + k] + 1e-5f) * gamma[k] + beta[k];
            acc += tn*trW[k*TE_ + c];
        }
    }
    x[(long)tok*C_ + col] = acc;
}

// ------------------------------------------------------------------
// row layernorm: x fp32 [rows][C] -> h bf16
// ------------------------------------------------------------------
__global__ void ln_kernel(const float* __restrict__ x, const float* __restrict__ w,
                          const float* __restrict__ b, unsigned short* __restrict__ hbf) {
    int row = blockIdx.x, tid = threadIdx.x;
    const float* xr = x + (long)row*C_;
    float v[4], s = 0.f, s2 = 0.f;
    #pragma unroll
    for (int i = 0; i < 4; i++) { v[i] = xr[tid + 256*i]; s += v[i]; s2 += v[i]*v[i]; }
    __shared__ float sh[512];
    sh[tid] = s; sh[256 + tid] = s2; __syncthreads();
    for (int o = 128; o > 0; o >>= 1) {
        if (tid < o) { sh[tid] += sh[tid + o]; sh[256 + tid] += sh[256 + tid + o]; }
        __syncthreads();
    }
    float m  = sh[0] / C_;
    float rs = rsqrtf(sh[256] / C_ - m*m + 1e-5f);
    #pragma unroll
    for (int i = 0; i < 4; i++) {
        int c = tid + 256*i;
        hbf[(long)row*C_ + c] = f2bf((v[i] - m)*rs*w[c] + b[c]);
    }
}

// ------------------------------------------------------------------
// weight stage: W fp32 [K][N] -> Wt bf16 [N][K]  (LDS tile transpose)
// ------------------------------------------------------------------
__global__ void wtrans_kernel(const float* __restrict__ W, unsigned short* __restrict__ Wt,
                              int K, int N) {
    __shared__ float tile[32][33];
    int k0 = blockIdx.y*32, n0 = blockIdx.x*32;
    int tx = threadIdx.x, ty = threadIdx.y;   // 32 x 8
    #pragma unroll
    for (int j = 0; j < 32; j += 8) tile[ty + j][tx] = W[(long)(k0 + ty + j)*N + n0 + tx];
    __syncthreads();
    #pragma unroll
    for (int j = 0; j < 32; j += 8) Wt[(long)(n0 + ty + j)*K + k0 + tx] = f2bf(tile[tx][ty + j]);
}

// ------------------------------------------------------------------
// bf16 WMMA GEMM:  out[M,N] = A[M,K](bf16) * Wt[N,K](bf16)^T + bias
// block tile 128x128, 8 waves (4x2), wave tile 32x64 (2x4 WMMA tiles),
// K-step 32; double-buffered LDS fed by async global->LDS copies.
// ------------------------------------------------------------------
#define GEMM_OUT_F32  1
#define GEMM_OUT_BF16 2
#define GEMM_RELU     4
#define GEMM_RESID    8

__global__ __launch_bounds__(256) void gemm_kernel(
    const unsigned short* __restrict__ A, const unsigned short* __restrict__ Wt,
    int M, int N, int K,
    const float* __restrict__ bias, const float* __restrict__ resid,
    float* __restrict__ outf, unsigned short* __restrict__ outb, int flags)
{
    __shared__ __align__(16) unsigned short ldsA[2][128*32];  // 2 x 8 KB
    __shared__ __align__(16) unsigned short ldsB[2][128*32];  // 2 x 8 KB
    int tid  = threadIdx.x;
    int wave = tid >> 5, lane = tid & 31;
    int wr = wave & 3, wc = wave >> 2;          // 4x2 wave grid
    int lhalf = lane >> 4, l15 = lane & 15;
    int bm = blockIdx.x*128, bn = blockIdx.y*128;

    // per-thread staging coordinates (2 x 16B for A, 2 x 16B for B)
    int r0 = tid >> 2, s0 = tid & 3;            // i = tid
    int r1 = (tid + 256) >> 2, s1 = tid & 3;    // i = tid + 256

    v8f acc[2][4];
    #pragma unroll
    for (int i = 0; i < 2; i++)
        #pragma unroll
        for (int j = 0; j < 4; j++) acc[i][j] = vzero();

    // prologue: stage k0 = 0 into buffer 0
    {
        async_copy_b128(LDS_OFF(&ldsA[0][r0*32 + s0*8]), &A [(long)(bm + r0)*K + s0*8]);
        async_copy_b128(LDS_OFF(&ldsA[0][r1*32 + s1*8]), &A [(long)(bm + r1)*K + s1*8]);
        async_copy_b128(LDS_OFF(&ldsB[0][r0*32 + s0*8]), &Wt[(long)(bn + r0)*K + s0*8]);
        async_copy_b128(LDS_OFF(&ldsB[0][r1*32 + s1*8]), &Wt[(long)(bn + r1)*K + s1*8]);
        wait_async0();
        __syncthreads();
    }

    int buf = 0;
    for (int k0 = 0; k0 < K; k0 += 32) {
        // prefetch next K tile into the other buffer (no wait yet)
        if (k0 + 32 < K) {
            int nb = buf ^ 1, kn = k0 + 32;
            async_copy_b128(LDS_OFF(&ldsA[nb][r0*32 + s0*8]), &A [(long)(bm + r0)*K + kn + s0*8]);
            async_copy_b128(LDS_OFF(&ldsA[nb][r1*32 + s1*8]), &A [(long)(bm + r1)*K + kn + s1*8]);
            async_copy_b128(LDS_OFF(&ldsB[nb][r0*32 + s0*8]), &Wt[(long)(bn + r0)*K + kn + s0*8]);
            async_copy_b128(LDS_OFF(&ldsB[nb][r1*32 + s1*8]), &Wt[(long)(bn + r1)*K + kn + s1*8]);
        }

        Frag afr[2], bfr[4];
        const unsigned* uA = (const unsigned*)&ldsA[buf][0];   // [128][16] dwords
        const unsigned* uB = (const unsigned*)&ldsB[buf][0];   // [128][16] dwords
        #pragma unroll
        for (int sm = 0; sm < 2; sm++) {
            int m = wr*32 + sm*16 + l15;
            #pragma unroll
            for (int d = 0; d < 8; d++) {
                int kp = (d >> 2)*16 + lhalf*8 + (d & 3)*2;     // ISA 16-bit A layout
                afr[sm].u[d] = uA[m*16 + (kp >> 1)];
            }
        }
        #pragma unroll
        for (int sn = 0; sn < 4; sn++) {
            int n = wc*64 + sn*16 + l15;
            #pragma unroll
            for (int d = 0; d < 8; d++) {
                int ks = (d >> 2)*16 + lhalf*8 + (d & 3)*2;     // ISA 16-bit B layout
                bfr[sn].u[d] = uB[n*16 + (ks >> 1)];
            }
        }
        #pragma unroll
        for (int sm = 0; sm < 2; sm++)
            #pragma unroll
            for (int sn = 0; sn < 4; sn++)
                acc[sm][sn] = __builtin_amdgcn_wmma_f32_16x16x32_bf16(
                    false, afr[sm].v, false, bfr[sn].v, (short)0, acc[sm][sn], false, false);

        wait_async0();        // our prefetch into buf^1 landed
        __syncthreads();      // everyone done reading buf / writing buf^1
        buf ^= 1;
    }

    // epilogue
    #pragma unroll
    for (int sm = 0; sm < 2; sm++) {
        #pragma unroll
        for (int sn = 0; sn < 4; sn++) {
            int n = bn + wc*64 + sn*16 + l15;
            int mb = bm + wr*32 + sm*16 + 8*lhalf;
            float bv = bias ? bias[n] : 0.f;
            #pragma unroll
            for (int r = 0; r < 8; r++) {
                long idx = (long)(mb + r)*N + n;
                float v = acc[sm][sn][r] + bv;
                if (flags & GEMM_RELU)  v = fmaxf(v, 0.f);
                if (flags & GEMM_RESID) v += resid[idx];
                if (flags & GEMM_OUT_F32)  outf[idx] = v;
                if (flags & GEMM_OUT_BF16) outb[idx] = f2bf(v);
            }
        }
    }
}

// ------------------------------------------------------------------
// V transpose: qkv bf16 [tok][3C] -> vt bf16 [BH][DH][L]
// ------------------------------------------------------------------
__global__ void vtrans_kernel(const unsigned short* __restrict__ qkv,
                              unsigned short* __restrict__ vt) {
    long i = (long)blockIdx.x*256 + threadIdx.x;   // over 32*64*1024
    int seq = (int)(i & 1023);
    long rest = i >> 10;
    int dh = (int)(rest & 63);
    int bh = (int)(rest >> 6);
    int b = bh >> 4, h = bh & 15;
    vt[i] = qkv[((long)(b*L_ + seq))*3072 + 2048 + h*64 + dh];
}

// ------------------------------------------------------------------
// flash-style causal attention.  grid (B*H, L/64), 128 threads (4 waves).
// wave owns 16 query rows; 32-key steps; S via WMMA (Kdim=DH), online
// softmax with 16-lane shuffles, P through LDS to A-layout, O += P*V.
// ------------------------------------------------------------------
__global__ __launch_bounds__(128) void attn_kernel(const unsigned short* __restrict__ qkv,
                                                   const unsigned short* __restrict__ vt,
                                                   unsigned short* __restrict__ attn) {
    __shared__ unsigned Pu[4][16][16];   // per-wave 16x32 bf16 P tile
    int tid = threadIdx.x, wave = tid >> 5, lane = tid & 31;
    int lhalf = lane >> 4, l15 = lane & 15;
    int bh = blockIdx.x, b = bh >> 4, h = bh & 15;
    int q0 = blockIdx.y*64 + wave*16;

    Frag qf[2];
    {
        int m = q0 + l15;
        const unsigned short* qrow = qkv + ((long)(b*L_ + m))*3072 + h*64;
        #pragma unroll
        for (int c = 0; c < 2; c++)
            #pragma unroll
            for (int d = 0; d < 8; d++) {
                int dh = c*32 + lhalf*16 + 2*d;
                qf[c].u[d] = *(const unsigned*)(qrow + dh);
            }
    }

    v8f o[4];
    #pragma unroll
    for (int t = 0; t < 4; t++) o[t] = vzero();
    float mi[8], li[8];
    #pragma unroll
    for (int r = 0; r < 8; r++) { mi[r] = -1e30f; li[r] = 0.f; }

    for (int kb = 0; kb < q0 + 16; kb += 32) {
        v8f s[2];
        #pragma unroll
        for (int t2 = 0; t2 < 2; t2++) {
            Frag kf0, kf1;
            int key = kb + t2*16 + l15;
            const unsigned short* krow = qkv + ((long)(b*L_ + key))*3072 + 1024 + h*64;
            #pragma unroll
            for (int d = 0; d < 8; d++) {
                kf0.u[d] = *(const unsigned*)(krow +      lhalf*16 + 2*d);
                kf1.u[d] = *(const unsigned*)(krow + 32 + lhalf*16 + 2*d);
            }
            v8f z = vzero();
            z = __builtin_amdgcn_wmma_f32_16x16x32_bf16(false, qf[0].v, false, kf0.v, (short)0, z, false, false);
            z = __builtin_amdgcn_wmma_f32_16x16x32_bf16(false, qf[1].v, false, kf1.v, (short)0, z, false, false);
            s[t2] = z;
        }

        unsigned short* Pl = (unsigned short*)&Pu[wave][0][0];
        #pragma unroll
        for (int r = 0; r < 8; r++) {
            int q = q0 + r + 8*lhalf;
            float s0 = s[0][r]*0.125f; if (kb      + l15 > q) s0 = -1e30f;
            float s1 = s[1][r]*0.125f; if (kb + 16 + l15 > q) s1 = -1e30f;
            float rm = fmaxf(s0, s1);
            #pragma unroll
            for (int off = 1; off < 16; off <<= 1) rm = fmaxf(rm, __shfl_xor(rm, off));
            float mn = fmaxf(mi[r], rm);
            float alpha = __expf(mi[r] - mn);
            float p0 = __expf(s0 - mn), p1 = __expf(s1 - mn);
            float rs = p0 + p1;
            #pragma unroll
            for (int off = 1; off < 16; off <<= 1) rs += __shfl_xor(rs, off);
            li[r] = li[r]*alpha + rs;
            mi[r] = mn;
            #pragma unroll
            for (int t = 0; t < 4; t++) o[t][r] *= alpha;
            int prow = r + 8*lhalf;
            Pl[prow*32 +      l15] = f2bf(p0);
            Pl[prow*32 + 16 + l15] = f2bf(p1);
        }
        asm volatile("s_wait_dscnt 0" ::: "memory");   // per-wave LDS RAW fence

        Frag pf;
        {
            const unsigned* up = &Pu[wave][0][0];
            #pragma unroll
            for (int d = 0; d < 8; d++) {
                int kp = lhalf*16 + d*2;
                pf.u[d] = up[l15*16 + (kp >> 1)];
            }
        }
        #pragma unroll
        for (int t = 0; t < 4; t++) {
            Frag vf;
            const unsigned short* vrow = vt + ((long)bh*64 + t*16 + l15)*L_ + kb + lhalf*16;
            #pragma unroll
            for (int d = 0; d < 8; d++) vf.u[d] = *(const unsigned*)(vrow + 2*d);
            o[t] = __builtin_amdgcn_wmma_f32_16x16x32_bf16(false, pf.v, false, vf.v, (short)0, o[t], false, false);
        }
    }

    #pragma unroll
    for (int t = 0; t < 4; t++)
        #pragma unroll
        for (int r = 0; r < 8; r++) {
            int q = q0 + r + 8*lhalf;
            attn[((long)(b*L_ + q))*C_ + h*64 + t*16 + l15] = f2bf(o[t][r] / li[r]);
        }
}

// ------------------------------------------------------------------
__global__ void tanh_kernel(const float* __restrict__ x, float* __restrict__ out) {
    long i = (long)blockIdx.x*256 + threadIdx.x;
    out[i] = tanhf(x[i]);
}

// ------------------------------------------------------------------
extern "C" void kernel_launch(void* const* d_in, const int* in_sizes, int n_in,
                              void* d_out, int out_size, void* d_ws, size_t ws_size,
                              hipStream_t stream) {
    const float* obs   = (const float*)d_in[0];
    const float* act   = (const float*)d_in[1];
    const float* rew   = (const float*)d_in[2];
    const float* obsW  = (const float*)d_in[3];
    const float* obsb  = (const float*)d_in[4];
    const float* actW  = (const float*)d_in[5];
    const float* actb  = (const float*)d_in[6];
    const float* rewW  = (const float*)d_in[7];
    const float* rewb  = (const float*)d_in[8];
    const float* trW   = (const float*)d_in[9];
    const float* trb   = (const float*)d_in[10];
    const float* gamma = (const float*)d_in[11];
    const float* beta  = (const float*)d_in[12];
    const float* ln1w  = (const float*)d_in[13];
    const float* ln1b  = (const float*)d_in[14];
    const float* qkvW  = (const float*)d_in[15];
    const float* qkvb  = (const float*)d_in[16];
    const float* outW  = (const float*)d_in[17];
    const float* outb  = (const float*)d_in[18];
    const float* ln2w  = (const float*)d_in[19];
    const float* ln2b  = (const float*)d_in[20];
    const float* fcW   = (const float*)d_in[21];
    const float* fcb   = (const float*)d_in[22];
    const float* pjW   = (const float*)d_in[23];
    const float* pjb   = (const float*)d_in[24];

    // ---- carve workspace ----
    char* w = (char*)d_ws;
    size_t off = 0;
    auto carve = [&](size_t bytes) -> void* {
        void* p = w + off;
        off += (bytes + 255) & ~(size_t)255;
        return p;
    };
    float*          stats = (float*)carve(128*sizeof(float));
    float*          x     = (float*)carve((size_t)MTOK*C_*4);
    unsigned short* hbf   = (unsigned short*)carve((size_t)MTOK*C_*2);
    unsigned short* wT    = (unsigned short*)carve((size_t)FF_*C_*2);       // 8 MB stage
    unsigned short* qkvbf = (unsigned short*)carve((size_t)MTOK*3*C_*2);
    unsigned short* vtb   = (unsigned short*)carve((size_t)B_*H_*DH_*L_*2);
    unsigned short* attnb = (unsigned short*)carve((size_t)MTOK*C_*2);
    unsigned short* ffbf  = (unsigned short*)carve((size_t)MTOK*FF_*2);

    // ---- encoder ----
    bn_stats_kernel<<<64, 256, 0, stream>>>(obs, stats);
    encode_kernel<<<dim3(MTOK, 4), 256, 0, stream>>>(obs, act, rew, obsW, obsb, actW, actb,
                                                     rewW, rewb, trW, trb, gamma, beta, stats, x);

    dim3 tb(32, 8);
    for (int l = 0; l < NL_; l++) {
        // LN1 -> h
        ln_kernel<<<MTOK, 256, 0, stream>>>(x, ln1w + l*C_, ln1b + l*C_, hbf);
        // QKV = h @ Wqkv + b   (bf16 out)
        wtrans_kernel<<<dim3(3*C_/32, C_/32), tb, 0, stream>>>(qkvW + (size_t)l*C_*3*C_, wT, C_, 3*C_);
        gemm_kernel<<<dim3(MTOK/128, 3*C_/128), 256, 0, stream>>>(
            hbf, wT, MTOK, 3*C_, C_, qkvb + l*3*C_, nullptr, nullptr, qkvbf, GEMM_OUT_BF16);
        // V transpose + attention
        vtrans_kernel<<<(B_*H_*DH_*L_)/256, 256, 0, stream>>>(qkvbf, vtb);
        attn_kernel<<<dim3(B_*H_, L_/64), 128, 0, stream>>>(qkvbf, vtb, attnb);
        // x = x + attn @ Wout + b
        wtrans_kernel<<<dim3(C_/32, C_/32), tb, 0, stream>>>(outW + (size_t)l*C_*C_, wT, C_, C_);
        gemm_kernel<<<dim3(MTOK/128, C_/128), 256, 0, stream>>>(
            attnb, wT, MTOK, C_, C_, outb + l*C_, x, x, nullptr, GEMM_OUT_F32 | GEMM_RESID);
        // LN2 -> h
        ln_kernel<<<MTOK, 256, 0, stream>>>(x, ln2w + l*C_, ln2b + l*C_, hbf);
        // ff = relu(h @ Wfc + b)   (bf16 out)
        wtrans_kernel<<<dim3(FF_/32, C_/32), tb, 0, stream>>>(fcW + (size_t)l*C_*FF_, wT, C_, FF_);
        gemm_kernel<<<dim3(MTOK/128, FF_/128), 256, 0, stream>>>(
            hbf, wT, MTOK, FF_, C_, fcb + l*FF_, nullptr, nullptr, ffbf, GEMM_OUT_BF16 | GEMM_RELU);
        // x = x + ff @ Wpj + b
        wtrans_kernel<<<dim3(C_/32, FF_/32), tb, 0, stream>>>(pjW + (size_t)l*FF_*C_, wT, FF_, C_);
        gemm_kernel<<<dim3(MTOK/128, C_/128), 256, 0, stream>>>(
            ffbf, wT, MTOK, C_, FF_, pjb + l*C_, x, x, nullptr, GEMM_OUT_F32 | GEMM_RESID);
    }

    tanh_kernel<<<((long)MTOK*C_)/256, 256, 0, stream>>>(x, (float*)d_out);
}
